// VectorQuantizer_54314156425321
// MI455X (gfx1250) — matrix operand order repaired and checked by reading, not compiled, not run on hardware
//
#include <hip/hip_runtime.h>
#include <hip/hip_bf16.h>

typedef _Float16 v8h  __attribute__((ext_vector_type(8)));
typedef _Float16 v16h __attribute__((ext_vector_type(16)));
typedef float    v8f  __attribute__((ext_vector_type(8)));
typedef float    v4f  __attribute__((ext_vector_type(4)));

#define K_CB     8192
#define D_DIM    256
#define N_ROWS   32768
#define NTILES   (K_CB / 16)          // 512 codebook tiles of 16 entries
#define LDS_STR  264                  // padded row stride in halves (breaks bank conflicts)
#define W_SCALE  4096.0f              // lift codebook out of f16-subnormal range
#define INV2S    (2.0f / 4096.0f)

// ---- CDNA5 async memory->LDS copy (32B per lane: two B128 ops, IOFFSET hits
// both the LDS and the global address per ISA 08_async_tensor §4.4). Tracked
// by ASYNCcnt, drained with s_wait_asynccnt before the workgroup barrier.
static __device__ __forceinline__ void async_cp32(unsigned lds_byte_addr,
                                                  const _Float16* g) {
    asm volatile("global_load_async_to_lds_b128 %0, %1, off\n\t"
                 "global_load_async_to_lds_b128 %0, %1, off offset:16"
                 :
                 : "v"(lds_byte_addr), "v"(g)
                 : "memory");
}
static __device__ __forceinline__ void async_drain() {
    asm volatile("s_wait_asynccnt 0x0" ::: "memory");
}

// B fragment: K-halves bp+ks*32+0..7 and +16..23 (lane half picks K sel*8.., 16+sel*8..)
static __device__ __forceinline__ v16h loadB(const _Float16* bp, int ks) {
    const v8h b0 = *(const v8h*)(bp + ks * 32);
    const v8h b1 = *(const v8h*)(bp + ks * 32 + 16);
    v16h b;
    #pragma unroll
    for (int i = 0; i < 8; ++i) { b[i] = b0[i]; b[i + 8] = b1[i]; }
    return b;
}

// ---------------- init: zero histogram ----------------
__global__ void k_init(unsigned* __restrict__ counts) {
    int t = blockIdx.x * blockDim.x + threadIdx.x;
    if (t < K_CB) counts[t] = 0u;
}

// ---------------- codebook -> scaled f16, e_sq ----------------
__global__ void __launch_bounds__(256)
k_prep_w(const float* __restrict__ w, _Float16* __restrict__ wh,
         float* __restrict__ esq) {
    int row = blockIdx.x;            // one block per codebook entry
    int t   = threadIdx.x;           // 256 threads, one element each
    float v = w[(size_t)row * D_DIM + t];
    wh[(size_t)row * D_DIM + t] = (_Float16)(v * W_SCALE);
    __shared__ float red[256];
    red[t] = v * v;
    __syncthreads();
    for (int s = 128; s > 0; s >>= 1) {
        if (t < s) red[t] += red[t + s];
        __syncthreads();
    }
    if (t == 0) esq[row] = red[0];
}

// ---------------- fused GEMM + argmin ----------------
// block = 8 waves; wave w owns 32 rows (two 16-row M-tiles) so every B fragment
// staged through LDS feeds TWO v_wmma ops. Codebook tiles stream through a
// double-buffered LDS stage filled by GLOBAL_LOAD_ASYNC_TO_LDS_B128, and the
// B-fragment ds_loads are software-pipelined one step ahead of the WMMAs.
// score(row, e) = ||e||^2 - 2*z.e   (z_sq constant per row -> dropped)
__global__ void __launch_bounds__(256)
k_argmin(const float* __restrict__ z,
         const _Float16* __restrict__ wh,
         const float* __restrict__ esq,
         int*   __restrict__ idx_out,
         float* __restrict__ idx_out_f) {
    __shared__ __align__(16) _Float16 tile[2][16 * LDS_STR];

    const int tid  = threadIdx.x;
    const int lane = tid & 31;
    const int wave = tid >> 5;            // 0..7
    const int n    = lane & 15;           // A-row / B-col / C-col slot for this lane
    const int sel  = lane >> 4;           // lane half (K-split & C row offset)
    const int rowBase = blockIdx.x * 256 + wave * 32;

    // ---- A fragments: 2 x (16 rows x 256 depth), f32 -> f16 in registers ----
    v16h A0[8], A1[8];
    {
        const float* rp0 = z + (size_t)(rowBase + n) * D_DIM;
        const float* rp1 = z + (size_t)(rowBase + 16 + n) * D_DIM;
        #pragma unroll
        for (int ks = 0; ks < 8; ++ks) {
            const int kb = ks * 32;
            // ISA A layout: lane half sel -> K = kb+sel*8..+7 (V0-3), kb+16+sel*8..+7 (V4-7)
            const v8f f0 = *(const v8f*)(rp0 + kb + sel * 8);
            const v8f f1 = *(const v8f*)(rp0 + kb + 16 + sel * 8);
            const v8f g0 = *(const v8f*)(rp1 + kb + sel * 8);
            const v8f g1 = *(const v8f*)(rp1 + kb + 16 + sel * 8);
            v16h a0, a1;
            #pragma unroll
            for (int i = 0; i < 8; ++i) {
                a0[i] = (_Float16)f0[i];  a0[i + 8] = (_Float16)f1[i];
                a1[i] = (_Float16)g0[i];  a1[i + 8] = (_Float16)g1[i];
            }
            A0[ks] = a0;
            A1[ks] = a1;
        }
    }

    float best[16];
    int   bidx[16];
    #pragma unroll
    for (int r = 0; r < 16; ++r) { best[r] = 3.4e38f; bidx[r] = 0; }

    // staging role: thread copies 32B; entry se (0..15), chunk sc (0..15)
    const int se = tid >> 4;
    const int sc = tid & 15;
    const unsigned ldsDst0 = (unsigned)(size_t)(&tile[0][se * LDS_STR + sc * 16]);
    const unsigned ldsDst1 = (unsigned)(size_t)(&tile[1][se * LDS_STR + sc * 16]);
    const _Float16* gsrc   = wh + (size_t)se * D_DIM + sc * 16;

    // prologue: async-stage tile 0
    async_cp32(ldsDst0, gsrc);
    async_drain();
    __syncthreads();

    for (int t = 0; t < NTILES; ++t) {
        const int buf = t & 1;
        if (t + 1 < NTILES) {  // fire-and-forget DMA of the next tile into the other buffer
            async_cp32(buf ? ldsDst0 : ldsDst1,
                       gsrc + (size_t)(t + 1) * 16 * D_DIM);
        }

        // ---- compute two 16x16 score tiles sharing each B fragment;
        //      ds_loads for step ks+1 issue before the WMMAs consuming step ks ----
        v8f acc0 = {0.f, 0.f, 0.f, 0.f, 0.f, 0.f, 0.f, 0.f};
        v8f acc1 = {0.f, 0.f, 0.f, 0.f, 0.f, 0.f, 0.f, 0.f};
        const _Float16* bp = &tile[buf][n * LDS_STR + sel * 8];
        v16h bcur = loadB(bp, 0);
        #pragma unroll
        for (int ks = 0; ks < 8; ++ks) {
            v16h bnxt = bcur;
            if (ks < 7) bnxt = loadB(bp, ks + 1);
            acc0 = __builtin_amdgcn_wmma_f32_16x16x32_f16(
                false, A0[ks], false, bcur, (short)0, acc0, false, false);
            acc1 = __builtin_amdgcn_wmma_f32_16x16x32_f16(
                false, A1[ks], false, bcur, (short)0, acc1, false, false);
            bcur = bnxt;
        }

        // ---- fused argmin update: lane holds column e, rows r+8*sel ----
        const int   e  = t * 16 + n;
        const float ev = esq[e];
        #pragma unroll
        for (int r = 0; r < 8; ++r) {
            float s0 = fmaf(-INV2S, acc0[r], ev);   // e_sq - 2*z.e (descale WMMA result)
            float s1 = fmaf(-INV2S, acc1[r], ev);
            if (s0 < best[r])     { best[r]     = s0; bidx[r]     = e; }
            if (s1 < best[r + 8]) { best[r + 8] = s1; bidx[r + 8] = e; }
        }

        async_drain();        // our next-tile DMA is done (long since, behind the WMMAs)
        __syncthreads();      // all waves' DMAs visible before anyone reads the other buffer
    }

    // ---- cross-lane min+index over the 16 lanes of each half ----
    #pragma unroll
    for (int m = 1; m < 16; m <<= 1) {
        #pragma unroll
        for (int r = 0; r < 16; ++r) {
            float ob = __shfl_xor(best[r], m, 32);
            int   oi = __shfl_xor(bidx[r], m, 32);
            if (ob < best[r] || (ob == best[r] && oi < bidx[r])) {
                best[r] = ob; bidx[r] = oi;
            }
        }
    }
    if (n == 0) {
        #pragma unroll
        for (int r = 0; r < 8; ++r) {
            int row0 = rowBase + r + sel * 8;           // M-tile 0, hi half = rows 8..15
            int row1 = rowBase + 16 + r + sel * 8;      // M-tile 1
            idx_out[row0]   = bidx[r];
            idx_out_f[row0] = (float)bidx[r];
            idx_out[row1]   = bidx[r + 8];
            idx_out_f[row1] = (float)bidx[r + 8];
        }
    }
}

// ---------------- gather + straight-through output + loss partials ----------------
__global__ void __launch_bounds__(256)
k_gather(const float* __restrict__ z, const float* __restrict__ w,
         const int* __restrict__ idx, float* __restrict__ outq,
         float* __restrict__ partial) {
    int tid = blockIdx.x * 256 + threadIdx.x;   // N*D/4 threads
    int row = tid >> 6;
    int j   = (tid & 63) << 2;
    int e   = idx[row];
    v4f q  = *(const v4f*)(w + (size_t)e * D_DIM + j);
    v4f zz = *(const v4f*)(z + (size_t)row * D_DIM + j);
    v4f st;
    float s = 0.f;
    #pragma unroll
    for (int i = 0; i < 4; ++i) {
        float d = q[i] - zz[i];
        st[i] = zz[i] + d;                      // matches z + stopgrad(q - z) exactly
        s += d * d;
    }
    *(v4f*)(outq + (size_t)row * D_DIM + j) = st;

    __shared__ float red[256];
    red[threadIdx.x] = s;
    __syncthreads();
    for (int s2 = 128; s2 > 0; s2 >>= 1) {
        if (threadIdx.x < s2) red[threadIdx.x] += red[threadIdx.x + s2];
        __syncthreads();
    }
    if (threadIdx.x == 0) partial[blockIdx.x] = red[0];
}

// ---------------- histogram (integer atomics: deterministic) ----------------
__global__ void k_hist(const int* __restrict__ idx, unsigned* __restrict__ counts) {
    int t = blockIdx.x * 256 + threadIdx.x;
    if (t < N_ROWS) atomicAdd(&counts[idx[t]], 1u);
}

// ---------------- finalize: loss + perplexity (single block, fixed order) ----------------
__global__ void __launch_bounds__(256)
k_final(const float* __restrict__ partial, const unsigned* __restrict__ counts,
        float* __restrict__ out_loss, float* __restrict__ out_perp) {
    __shared__ float redL[256];
    __shared__ float redH[256];
    int t = threadIdx.x;
    float sl = 0.f, sh = 0.f;
    for (int i = 0; i < 32; ++i) sl += partial[t * 32 + i];        // 8192 partials
    for (int i = 0; i < 32; ++i) {                                 // 8192 bins
        float p = (float)counts[t * 32 + i] * (1.0f / (float)N_ROWS);
        sh -= p * logf(p + 1e-10f);
    }
    redL[t] = sl; redH[t] = sh;
    __syncthreads();
    for (int s = 128; s > 0; s >>= 1) {
        if (t < s) { redL[t] += redL[t + s]; redH[t] += redH[t + s]; }
        __syncthreads();
    }
    if (t == 0) {
        *out_loss = 1.25f * redL[0] / (float)(N_ROWS * D_DIM);     // q_loss + 0.25*e_loss
        *out_perp = expf(redH[0]);
    }
}

extern "C" void kernel_launch(void* const* d_in, const int* in_sizes, int n_in,
                              void* d_out, int out_size, void* d_ws, size_t ws_size,
                              hipStream_t stream) {
    const float* z = (const float*)d_in[0];     // [32,1024,256] f32
    const float* w = (const float*)d_in[1];     // [8192,256]    f32
    float* out = (float*)d_out;

    // d_out layout (tuple flattened): quantized_st[8388608], loss[1], perp[1], indices[32768]
    float* out_q    = out;
    float* out_loss = out + (size_t)N_ROWS * D_DIM;
    float* out_perp = out_loss + 1;
    float* out_idxf = out_perp + 1;

    // workspace layout
    char* ws = (char*)d_ws;
    _Float16* wh      = (_Float16*)(ws);                                   // 4 MiB
    float*    esq     = (float*)(ws + (size_t)K_CB * D_DIM * 2);           // 32 KiB
    unsigned* counts  = (unsigned*)((char*)esq + (size_t)K_CB * 4);        // 32 KiB
    int*      idx     = (int*)((char*)counts + (size_t)K_CB * 4);          // 128 KiB
    float*    partial = (float*)((char*)idx + (size_t)N_ROWS * 4);         // 32 KiB

    k_init  <<<K_CB / 256, 256, 0, stream>>>(counts);
    k_prep_w<<<K_CB,       256, 0, stream>>>(w, wh, esq);
    k_argmin<<<N_ROWS / 256, 256, 0, stream>>>(z, wh, esq, idx, out_idxf);
    k_gather<<<(N_ROWS * D_DIM / 4) / 256, 256, 0, stream>>>(z, w, idx, out_q, partial);
    k_hist  <<<N_ROWS / 256, 256, 0, stream>>>(idx, counts);
    k_final <<<1, 256, 0, stream>>>(partial, counts, out_loss, out_perp);
}
